// CausalSelfAttention_7241314861575
// MI455X (gfx1250) — matrix operand order repaired and checked by reading, not compiled, not run on hardware
//
#include <hip/hip_runtime.h>

// ---------------------------------------------------------------------------
// CDNA5 (gfx1250) causal self-attention pipeline, bf16 WMMA everywhere.
//   B=4, S=2048, D=1024, H=16, Hd=64
// GEMMs stage packed weight fragments through LDS with async global->LDS
// copies (double buffered), synchronized via s_wait_asynccnt + barriers.
// ---------------------------------------------------------------------------

#define BATCH   4
#define SEQ     2048
#define DMODEL  1024
#define NHEADS  16
#define HDIM    64
#define ROWS    (BATCH * SEQ)          // 8192

typedef __attribute__((ext_vector_type(16))) __bf16 v16bf;
typedef __attribute__((ext_vector_type(8)))  __bf16 v8bf;
typedef __attribute__((ext_vector_type(8)))  float  v8f;

#if defined(__has_builtin)
#  if __has_builtin(__builtin_amdgcn_global_load_async_to_lds_b128)
#    define HAVE_ASYNC_LDS 1
#  endif
#endif
#ifndef HAVE_ASYNC_LDS
#  define HAVE_ASYNC_LDS 0
#endif

// Builtin parameter types (from compiler diagnostic): pointers to a 16-byte
// int vector, source in the global (AS1) space, destination in LDS (AS3).
typedef int v4i_vs __attribute__((vector_size(16)));
typedef __attribute__((address_space(1))) v4i_vs* gvp_t;
typedef __attribute__((address_space(3))) v4i_vs* lvp_t;

static __device__ __forceinline__ void async_copy16(const __bf16* g, __bf16* l) {
#if HAVE_ASYNC_LDS
    __builtin_amdgcn_global_load_async_to_lds_b128((gvp_t)g, (lvp_t)l,
                                                   /*offset=*/0, /*cpol=*/0);
#else
    *(v8bf*)l = *(const v8bf*)g;
#endif
}

static __device__ __forceinline__ void wait_async_le2() {
#if HAVE_ASYNC_LDS
    asm volatile("s_wait_asynccnt 0x2" ::: "memory");
#endif
}
static __device__ __forceinline__ void wait_async_le0() {
#if HAVE_ASYNC_LDS
    asm volatile("s_wait_asynccnt 0x0" ::: "memory");
#endif
}

static __device__ __forceinline__ v8f vzero8() {
    v8f r;
#pragma unroll
    for (int i = 0; i < 8; ++i) r[i] = 0.0f;
    return r;
}

static __device__ __forceinline__ v16bf join8(v8bf lo, v8bf hi) {
    v16bf r;
#pragma unroll
    for (int i = 0; i < 8; ++i) { r[i] = lo[i]; r[8 + i] = hi[i]; }
    return r;
}

static __device__ __forceinline__ v8f wmma_bf16(v16bf a, v16bf b, v8f c) {
    return __builtin_amdgcn_wmma_f32_16x16x32_bf16(
        /*neg_a=*/false, a, /*neg_b=*/false, b,
        /*c_mod=*/(short)0, c, /*reuse_a=*/false, /*reuse_b=*/false);
}

// ---------------------------------------------------------------------------
// Kernel 1: f32 -> bf16 elementwise convert (activations).
// ---------------------------------------------------------------------------
__global__ void cvt_bf16_kernel(const float* __restrict__ src,
                                __bf16* __restrict__ dst, int n) {
    int i = blockIdx.x * 256 + threadIdx.x;
    if (i < n) dst[i] = (__bf16)src[i];
}

// ---------------------------------------------------------------------------
// Kernel 2: pack f32 weight [K x N] row-major into bf16 WMMA B-fragment order.
// Tile (kt, nt) covers K rows [kt*32, kt*32+32), cols [nt*16, nt*16+16).
// Packed element p = tile*512 + lane*16 + e holds
//   w[(kt*32 + (lane<16 ? 0:16) + e) * N + nt*16 + (lane&15)]
// so a lane's 16-element fragment is one contiguous 32-byte load.
// ---------------------------------------------------------------------------
__global__ void pack_w_kernel(const float* __restrict__ w,
                              __bf16* __restrict__ wp, int K, int N) {
    int p    = blockIdx.x * 256 + threadIdx.x;
    int e    = p & 15;
    int lane = (p >> 4) & 31;
    int tile = p >> 9;
    int ntiles = N >> 4;
    int nt = tile % ntiles;
    int kt = tile / ntiles;
    int krow = kt * 32 + ((lane < 16) ? 0 : 16) + e;
    int col  = nt * 16 + (lane & 15);
    wp[p] = (__bf16)w[krow * N + col];
}

// ---------------------------------------------------------------------------
// Shared GEMM core.  Block = 256 threads (8 waves) computing a 64x128 tile.
// Per K-step the block's B data is one contiguous 8KB chunk of packed
// fragments; it is staged into LDS with async global->LDS b128 copies,
// double buffered.  A rows are loaded directly (streamed once per block).
// ---------------------------------------------------------------------------
template <int N>
static __device__ __forceinline__ void gemm_core(const __bf16* __restrict__ A,
                                                 const __bf16* __restrict__ Wp,
                                                 __bf16 (&bsh)[2][4096],
                                                 v8f (&acc)[4],
                                                 int M0, int ntB, int wn) {
    const int K = DMODEL;
    const int NSTEPS = K / 32;
    int tid  = threadIdx.x;
    int lane = tid & 31;
    int l15  = lane & 15;
    int koff = (lane < 16) ? 0 : 16;

    const __bf16* arow = A + (size_t)(M0 + l15) * K;

    auto issue = [&](int kt, int buf) {
        const __bf16* src =
            Wp + (((size_t)kt * (N >> 4) + ntB) << 9) + tid * 16;
        __bf16* dst = &bsh[buf][tid * 16];
        async_copy16(src, dst);
        async_copy16(src + 8, dst + 8);
    };

    issue(0, 0);
    for (int kt = 0; kt < NSTEPS; ++kt) {
        int buf = kt & 1;
        if (kt + 1 < NSTEPS) {
            issue(kt + 1, buf ^ 1);   // prefetch next chunk into other buffer
            wait_async_le2();         // oldest 2 (current buffer) retired
        } else {
            wait_async_le0();
        }
        __syncthreads();              // all waves' fills visible

        int k0 = kt * 32 + koff;
        v16bf af = join8(*(const v8bf*)(arow + k0),
                         *(const v8bf*)(arow + k0 + 8));
        __builtin_prefetch(arow + k0 + 64, 0, 1);
#pragma unroll
        for (int ns = 0; ns < 4; ++ns) {
            v16bf bf = *(const v16bf*)&bsh[buf][(wn * 4 + ns) * 512 + lane * 16];
            acc[ns] = wmma_bf16(af, bf, acc[ns]);
        }
        __syncthreads();              // done reading before buffer reuse
    }
}

// ---------------------------------------------------------------------------
// Kernel 3: QKV projection GEMM.  A = xb [8192 x 1024] bf16 row-major,
// Wp = packed w_qkv fragments (K=1024, N=3072).
// Epilogue scatters: Q (scaled 1/8) and K as [B,H,S,Hd] bf16,
//                    V transposed as [B,H,Hd,S] bf16.
// ---------------------------------------------------------------------------
__global__ void gemm_qkv_kernel(const __bf16* __restrict__ A,
                                const __bf16* __restrict__ Wp,
                                __bf16* __restrict__ qb,
                                __bf16* __restrict__ kb,
                                __bf16* __restrict__ vtb) {
    __shared__ __align__(32) __bf16 bsh[2][4096];
    int lane = threadIdx.x & 31;
    int w    = threadIdx.x >> 5;
    int wm   = w & 3, wn = w >> 2;
    int M0   = blockIdx.x * 64 + wm * 16;
    int N0   = blockIdx.y * 128 + wn * 64;
    int l15  = lane & 15;

    v8f acc[4];
#pragma unroll
    for (int i = 0; i < 4; ++i) acc[i] = vzero8();

    gemm_core<3 * DMODEL>(A, Wp, bsh, acc, M0, blockIdx.y * 8, wn);

    int half8 = (lane < 16) ? 0 : 8;
#pragma unroll
    for (int ns = 0; ns < 4; ++ns) {
        int c   = N0 + ns * 16 + l15;
        int t   = c >> 10;          // 0=q,1=k,2=v
        int rem = c & 1023;
        int h   = rem >> 6;
        int hd  = rem & 63;
#pragma unroll
        for (int v = 0; v < 8; ++v) {
            int m = M0 + half8 + v;
            int b = m >> 11;        // /SEQ
            int s = m & 2047;
            float val = acc[ns][v];
            int bh = b * NHEADS + h;
            if (t == 0)
                qb[((size_t)(bh * SEQ + s) << 6) + hd] = (__bf16)(val * 0.125f);
            else if (t == 1)
                kb[((size_t)(bh * SEQ + s) << 6) + hd] = (__bf16)val;
            else
                vtb[((size_t)(bh * HDIM + hd) << 11) + s] = (__bf16)val;
        }
    }
}

// ---------------------------------------------------------------------------
// Kernel 4: flash attention.  grid = (SEQ/128, BATCH*NHEADS), block = 256.
// One wave owns 16 query rows; streams keys in 32-wide tiles with online
// softmax; P is converted C/D-layout -> A-layout through per-wave LDS.
// Output written bf16 into yb [8192 x 1024] (heads re-interleaved).
// ---------------------------------------------------------------------------
__global__ void flash_attn_kernel(const __bf16* __restrict__ qb,
                                  const __bf16* __restrict__ kb,
                                  const __bf16* __restrict__ vtb,
                                  __bf16* __restrict__ yb) {
    __shared__ __align__(32) __bf16 psh[8][16 * 32];

    int lane  = threadIdx.x & 31;
    int w     = threadIdx.x >> 5;
    int by    = blockIdx.y;                 // b*NHEADS + h
    int q0    = blockIdx.x * 128 + w * 16;
    int l15   = lane & 15;
    int koff8  = (lane < 16) ? 0 : 8;
    int koff16 = (lane < 16) ? 0 : 16;
    int half8  = (lane < 16) ? 0 : 8;

    const __bf16* qh = qb  + ((size_t)by << 17);   // *SEQ*HDIM
    const __bf16* kh = kb  + ((size_t)by << 17);
    const __bf16* vh = vtb + ((size_t)by << 17);

    // Q fragments (K = 64 -> two 16x32 A fragments), resident all loop.
    v16bf qf[2];
#pragma unroll
    for (int c = 0; c < 2; ++c) {
        const __bf16* p = qh + (size_t)(q0 + l15) * HDIM + c * 32 + koff16;
        qf[c] = join8(*(const v8bf*)p, *(const v8bf*)(p + 8));
    }

    v8f o0 = vzero8(), o1 = vzero8(), o2 = vzero8(), o3 = vzero8();
    float mi[8], li[8];
#pragma unroll
    for (int v = 0; v < 8; ++v) { mi[v] = -1.0e30f; li[v] = 0.0f; }

    for (int j0 = 0; j0 <= q0 + 15; j0 += 32) {
        // ---- S = Q @ K^T : two 16x16 f32 tiles (cols j0.. / j0+16..) ----
        v8f s0 = vzero8(), s1 = vzero8();
#pragma unroll
        for (int c = 0; c < 2; ++c) {
            v16bf b0 = *(const v16bf*)(kh + (size_t)(j0 + l15) * HDIM +
                                       c * 32 + koff16);
            v16bf b1 = *(const v16bf*)(kh + (size_t)(j0 + 16 + l15) * HDIM +
                                       c * 32 + koff16);
            s0 = wmma_bf16(qf[c], b0, s0);
            s1 = wmma_bf16(qf[c], b1, s1);
        }

        // ---- causal mask + online softmax ----
        int col0 = j0 + l15, col1 = col0 + 16;
        int rowb = q0 + half8;
#pragma unroll
        for (int v = 0; v < 8; ++v) {
            int r = rowb + v;
            if (col0 > r) s0[v] = -1.0e30f;
            if (col1 > r) s1[v] = -1.0e30f;
        }
#pragma unroll
        for (int v = 0; v < 8; ++v) {
            float mx = fmaxf(s0[v], s1[v]);
            for (int off = 1; off < 16; off <<= 1)
                mx = fmaxf(mx, __shfl_xor(mx, off, 32));
            float nm = fmaxf(mi[v], mx);
            float sc = __expf(mi[v] - nm);
            float p0 = __expf(s0[v] - nm);
            float p1 = __expf(s1[v] - nm);
            s0[v] = p0; s1[v] = p1;
            float sum = p0 + p1;
            for (int off = 1; off < 16; off <<= 1)
                sum += __shfl_xor(sum, off, 32);
            li[v] = li[v] * sc + sum;
            mi[v] = nm;
            o0[v] *= sc; o1[v] *= sc; o2[v] *= sc; o3[v] *= sc;
        }

        // ---- P (C/D layout) -> LDS -> A-fragment layout ----
        __bf16* pp = &psh[w][0];
#pragma unroll
        for (int v = 0; v < 8; ++v) {
            pp[(half8 + v) * 32 + l15]      = (__bf16)s0[v];
            pp[(half8 + v) * 32 + 16 + l15] = (__bf16)s1[v];
        }
        asm volatile("s_wait_dscnt 0" ::: "memory");
        const __bf16* pr = pp + l15 * 32 + koff16;
        v16bf pf = join8(*(const v8bf*)pr, *(const v8bf*)(pr + 8));

        // ---- O += P @ V (V stored [Hd][S], contiguous key runs) ----
        v16bf vf;
        vf = *(const v16bf*)(vh + ((size_t)(0 * 16 + l15) << 11) + j0 + koff16);
        o0 = wmma_bf16(pf, vf, o0);
        vf = *(const v16bf*)(vh + ((size_t)(1 * 16 + l15) << 11) + j0 + koff16);
        o1 = wmma_bf16(pf, vf, o1);
        vf = *(const v16bf*)(vh + ((size_t)(2 * 16 + l15) << 11) + j0 + koff16);
        o2 = wmma_bf16(pf, vf, o2);
        vf = *(const v16bf*)(vh + ((size_t)(3 * 16 + l15) << 11) + j0 + koff16);
        o3 = wmma_bf16(pf, vf, o3);
    }

    // ---- epilogue: normalize, re-interleave heads into [B*S, D] bf16 ----
    int b = by >> 4, h = by & 15;
#pragma unroll
    for (int v = 0; v < 8; ++v) {
        float inv = 1.0f / li[v];
        size_t gr = (size_t)(b * SEQ + q0 + half8 + v) * DMODEL;
        int colb = h * HDIM + l15;
        yb[gr + colb +  0] = (__bf16)(o0[v] * inv);
        yb[gr + colb + 16] = (__bf16)(o1[v] * inv);
        yb[gr + colb + 32] = (__bf16)(o2[v] * inv);
        yb[gr + colb + 48] = (__bf16)(o3[v] * inv);
    }
}

// ---------------------------------------------------------------------------
// Kernel 5: output projection GEMM.  yb [8192 x 1024] bf16 @ packed w_proj
// (K=1024, N=1024) -> f32 out [8192 x 1024].
// ---------------------------------------------------------------------------
__global__ void gemm_proj_kernel(const __bf16* __restrict__ A,
                                 const __bf16* __restrict__ Wp,
                                 float* __restrict__ out) {
    __shared__ __align__(32) __bf16 bsh[2][4096];
    const int N = DMODEL;
    int lane = threadIdx.x & 31;
    int w    = threadIdx.x >> 5;
    int wm   = w & 3, wn = w >> 2;
    int M0   = blockIdx.x * 64 + wm * 16;
    int N0   = blockIdx.y * 128 + wn * 64;
    int l15  = lane & 15;

    v8f acc[4];
#pragma unroll
    for (int i = 0; i < 4; ++i) acc[i] = vzero8();

    gemm_core<DMODEL>(A, Wp, bsh, acc, M0, blockIdx.y * 8, wn);

    int half8 = (lane < 16) ? 0 : 8;
#pragma unroll
    for (int ns = 0; ns < 4; ++ns) {
        int c = N0 + ns * 16 + l15;
#pragma unroll
        for (int v = 0; v < 8; ++v) {
            int m = M0 + half8 + v;
            out[(size_t)m * N + c] = acc[ns][v];
        }
    }
}

// ---------------------------------------------------------------------------
// Host launcher.
// ---------------------------------------------------------------------------
extern "C" void kernel_launch(void* const* d_in, const int* in_sizes, int n_in,
                              void* d_out, int out_size, void* d_ws, size_t ws_size,
                              hipStream_t stream) {
    const float* x     = (const float*)d_in[0];   // [4,2048,1024]
    const float* wqkv  = (const float*)d_in[1];   // [1024,3072]
    const float* wproj = (const float*)d_in[2];   // [1024,1024]
    float* out = (float*)d_out;                   // [4,2048,1024]

    char* ws = (char*)d_ws;
    size_t off = 0;
    auto alloc = [&](size_t bytes) {
        void* p = ws + off;
        off += (bytes + 255) & ~(size_t)255;
        return p;
    };

    __bf16* xb     = (__bf16*)alloc((size_t)ROWS * DMODEL * 2);        // 16 MB
    __bf16* wqkvp  = (__bf16*)alloc((size_t)DMODEL * 3 * DMODEL * 2);  //  6 MB
    __bf16* wprojp = (__bf16*)alloc((size_t)DMODEL * DMODEL * 2);      //  2 MB
    __bf16* qb     = (__bf16*)alloc((size_t)ROWS * DMODEL * 2);        // 16 MB
    __bf16* kb     = (__bf16*)alloc((size_t)ROWS * DMODEL * 2);        // 16 MB
    __bf16* vtb    = (__bf16*)alloc((size_t)ROWS * DMODEL * 2);        // 16 MB
    __bf16* yb     = (__bf16*)alloc((size_t)ROWS * DMODEL * 2);        // 16 MB

    int nx = ROWS * DMODEL;                                            // 8388608
    cvt_bf16_kernel<<<nx / 256, 256, 0, stream>>>(x, xb, nx);
    pack_w_kernel<<<(DMODEL * 3 * DMODEL) / 256, 256, 0, stream>>>(
        wqkv, wqkvp, DMODEL, 3 * DMODEL);
    pack_w_kernel<<<(DMODEL * DMODEL) / 256, 256, 0, stream>>>(
        wproj, wprojp, DMODEL, DMODEL);

    gemm_qkv_kernel<<<dim3(ROWS / 64, (3 * DMODEL) / 128), 256, 0, stream>>>(
        xb, wqkvp, qb, kb, vtb);

    flash_attn_kernel<<<dim3(SEQ / 128, BATCH * NHEADS), 256, 0, stream>>>(
        qb, kb, vtb, yb);

    gemm_proj_kernel<<<dim3(ROWS / 64, DMODEL / 128), 256, 0, stream>>>(
        yb, wprojp, out);
}
